// FocusedLinearAttentionV2_29033978921045
// MI455X (gfx1250) — matrix operand order repaired and checked by reading, not compiled, not run on hardware
//
#include <hip/hip_runtime.h>
#include <cstddef>

typedef __attribute__((ext_vector_type(16))) __bf16 v16bf;
typedef __attribute__((ext_vector_type(8)))  __bf16 v8bf;
typedef __attribute__((ext_vector_type(8)))  float  v8f;

#define FLA_N 36
#define FLA_C 128
#define EPS_F 1e-4f

// d_ws layout (bf16, weights transposed to [n][k], k contiguous):
//   qkw_t : elems [0,      32768)   (256 cols x 128)
//   vw_t  : elems [32768,  49152)   (128 cols x 128)
//   pw_t  : elems [49152,  65536)   (128 cols x 128)
__global__ void fla_prep_weights(const float* __restrict__ qkw,
                                 const float* __restrict__ vw,
                                 const float* __restrict__ pw,
                                 __bf16* __restrict__ ws) {
  int i = blockIdx.x * blockDim.x + threadIdx.x;
  if (i < 32768) {
    int n = i >> 7, k = i & 127;
    ws[i] = (__bf16)qkw[k * 256 + n];
  } else if (i < 49152) {
    int j = i - 32768; int n = j >> 7, k = j & 127;
    ws[i] = (__bf16)vw[k * 128 + n];
  } else if (i < 65536) {
    int j = i - 49152; int n = j >> 7, k = j & 127;
    ws[i] = (__bf16)pw[k * 128 + n];
  }
}

// A-fragment for v_wmma_f32_16x16x32_bf16:
// lane (m = L%16, g = L/16) holds K in [16g,16g+16) of the K32 slice.
__device__ __forceinline__ v16bf make_afrag(const __bf16* row_k32, int g) {
  const v8bf alo = *(const v8bf*)(row_k32 + g * 16);
  const v8bf ahi = *(const v8bf*)(row_k32 + g * 16 + 8);
  v16bf a;
#pragma unroll
  for (int i = 0; i < 8; i++) { a[i] = alo[i]; a[i + 8] = ahi[i]; }
  return a;
}

__global__ void __launch_bounds__(256)
fla_kernel(const float* __restrict__ x, const float* __restrict__ pos,
           const float* __restrict__ mask,
           const float* __restrict__ qk_b, const float* __restrict__ v_b,
           const float* __restrict__ proj_b,
           const long long* __restrict__ voxel_inds,
           const __bf16* __restrict__ wbf,
           float* __restrict__ out) {
  const int b    = blockIdx.x;
  const int tid  = threadIdx.x;
  const int wave = tid >> 5;
  const int lane = tid & 31;
  const int l16  = lane & 15;
  const int g    = lane >> 4;

  // ---- LDS carving (<= 61.6 KB static), with phase-safe aliasing ----
  __shared__ __align__(16) unsigned char smem[61600];
  __bf16* xp_lds  = (__bf16*)(smem);             // 48x128 bf16 (12288 B)   phase 0-1
  __bf16* x_lds   = (__bf16*)(smem + 12288);     // 48x128 bf16 (12288 B)   phase 0-1
  float*  q_lds   = (float*)(smem);              // 36x128 f32  (18432 B)   phase 1b-4 (aliases xp+x)
  float*  k_lds   = (float*)(smem + 24576);      // 36x128 f32  (18432 B)   phase 1b-3
  __bf16* out_lds = (__bf16*)(smem + 24576);     // 48x128 bf16 (12288 B)   phase 4-5 (aliases k)
  float*  v_lds   = (float*)(smem + 43008);      // 36x128 f32  (18432 B)   phase 1b-3
  float*  kv_lds  = (float*)(smem + 43008);      // 8x16x16 f32 (8192 B)    phase 3b-4 (aliases v)
  float*  mask_lds= (float*)(smem + 61440);      // 36 f32
  float*  num_lds = (float*)(smem + 61584);      // 1 f32

  const __bf16* w_qk   = wbf;
  const __bf16* w_v    = wbf + 32768;
  const __bf16* w_proj = wbf + 49152;

  // ---------------- Phase 0: load x/pos, build bf16 xp & x tiles ----------------
  const float* xg = x   + (size_t)b * FLA_N * FLA_C;
  const float* pg = pos + (size_t)b * FLA_N * FLA_C;
  for (int i = tid; i < (FLA_N * FLA_C) / 4; i += 256) {   // 1152 float4
    float4 xv = ((const float4*)xg)[i];
    float4 pv = ((const float4*)pg)[i];
    int o = i * 4;
    x_lds[o + 0] = (__bf16)xv.x;  xp_lds[o + 0] = (__bf16)(xv.x + pv.x);
    x_lds[o + 1] = (__bf16)xv.y;  xp_lds[o + 1] = (__bf16)(xv.y + pv.y);
    x_lds[o + 2] = (__bf16)xv.z;  xp_lds[o + 2] = (__bf16)(xv.z + pv.z);
    x_lds[o + 3] = (__bf16)xv.w;  xp_lds[o + 3] = (__bf16)(xv.w + pv.w);
  }
  // zero padding rows 36..47 of both A tiles
  for (int i = tid; i < 12 * FLA_C; i += 256) {
    xp_lds[FLA_N * FLA_C + i] = (__bf16)0.f;
    x_lds[FLA_N * FLA_C + i]  = (__bf16)0.f;
  }
  if (tid < FLA_N) mask_lds[tid] = mask[(size_t)b * FLA_N + tid];
  __syncthreads();

  if (tid == 0) {
    float s = 0.f;
    for (int n = 0; n < FLA_N; n++) s += mask_lds[n];
    num_lds[0] = s;
  }

  // ---------------- Phase 1: GEMM1 (qk) + GEMM2 (v) via WMMA ----------------
  // 72 tile-jobs: 0..47 -> qk (16 ntiles x 3 mtiles), 48..71 -> v (8 x 3). 9 per wave.
  v8f acc[9];
#pragma unroll
  for (int t = 0; t < 9; t++) {
    const int job   = wave + t * 8;
    const bool isqk = (job < 48);
    const int loc   = isqk ? job : (job - 48);
    const int nt    = isqk ? (loc & 15) : (loc & 7);
    const int mt    = isqk ? (loc >> 4) : (loc >> 3);
    const __bf16* A = isqk ? xp_lds : x_lds;
    const __bf16* W = isqk ? w_qk : w_v;
    const __bf16* arow = A + (mt * 16 + l16) * FLA_C;
    const __bf16* wp   = W + (nt * 16 + l16) * FLA_C + g * 16;
    v8f c = {0.f, 0.f, 0.f, 0.f, 0.f, 0.f, 0.f, 0.f};
#pragma unroll
    for (int kb = 0; kb < 4; kb++) {
      v16bf af = make_afrag(arow + kb * 32, g);
      v16bf bf = *(const v16bf*)(wp + kb * 32);
      c = __builtin_amdgcn_wmma_f32_16x16x32_bf16(false, af, false, bf,
                                                  (short)0, c, false, false);
    }
    acc[t] = c;
  }
  __syncthreads();   // all reads of xp/x done -> q may overwrite that region

  // ---------------- Phase 1b: bias + mask, spill q/k/v to LDS ----------------
#pragma unroll
  for (int t = 0; t < 9; t++) {
    const int job   = wave + t * 8;
    const bool isqk = (job < 48);
    const int loc   = isqk ? job : (job - 48);
    const int nt    = isqk ? (loc & 15) : (loc & 7);
    const int mt    = isqk ? (loc >> 4) : (loc >> 3);
    const int col   = nt * 16 + l16;
    const float bias = isqk ? qk_b[col] : v_b[col];
#pragma unroll
    for (int r = 0; r < 8; r++) {
      const int m = mt * 16 + g * 8 + r;
      if (m < FLA_N) {
        const float val = acc[t][r] + bias;
        if (isqk) {
          if (col < FLA_C) q_lds[m * FLA_C + col] = val;
          else             k_lds[m * FLA_C + (col - FLA_C)] = val * mask_lds[m];
        } else {
          v_lds[m * FLA_C + col] = val;
        }
      }
    }
  }
  __syncthreads();

  // ---------------- Phase 2: cube + normalize (q per-row/head, k per-column) ----------------
  for (int i = tid; i < FLA_N * 8; i += 256) {   // 288 head-rows
    const int n = i >> 3, h = i & 7;
    float* p = &q_lds[n * FLA_C + h * 16];
    float tv[16]; float ss = 0.f;
#pragma unroll
    for (int j = 0; j < 16; j++) { float qv = p[j]; qv = qv * qv * qv; tv[j] = qv; ss += qv * qv; }
    const float inv = 1.f / (EPS_F + sqrtf(ss));
#pragma unroll
    for (int j = 0; j < 16; j++) p[j] = tv[j] * inv;
  }
  if (tid < FLA_C) {                              // 128 columns, norm over n
    float ss = 0.f;
    for (int n = 0; n < FLA_N; n++) {
      float kv = k_lds[n * FLA_C + tid];
      kv = kv * kv * kv;
      k_lds[n * FLA_C + tid] = kv;
      ss += kv * kv;
    }
    const float inv = 1.f / (EPS_F + sqrtf(ss));
    for (int n = 0; n < FLA_N; n++) k_lds[n * FLA_C + tid] *= inv;
  }
  __syncthreads();

  // ---------------- Phase 3: kv[h][c][d] = sum_n k[n][h,c] * v[n][h,d] ----------------
  float kvr[8];
#pragma unroll
  for (int j = 0; j < 8; j++) {
    const int o = tid + 256 * j;                  // 2048 elements
    const int h = o >> 8, cc = (o >> 4) & 15, d = o & 15;
    float s = 0.f;
    for (int n = 0; n < FLA_N; n++)
      s += k_lds[n * FLA_C + h * 16 + cc] * v_lds[n * FLA_C + h * 16 + d];
    kvr[j] = s;
  }
  __syncthreads();   // v reads done -> kv may overwrite v region
#pragma unroll
  for (int j = 0; j < 8; j++) kv_lds[tid + 256 * j] = kvr[j];
  __syncthreads();

  // ---------------- Phase 4: out = (q @ kv) / (num+eps), bf16 into out_lds ----------------
  const float rnum = 1.f / (num_lds[0] + EPS_F);
  for (int i = tid; i < 12 * FLA_C; i += 256)     // zero padding rows 36..47
    out_lds[FLA_N * FLA_C + i] = (__bf16)0.f;
  for (int i = tid; i < FLA_N * FLA_C; i += 256) {
    const int n = i >> 7, cc = i & 127, h = cc >> 4, d = cc & 15;
    float s = 0.f;
#pragma unroll
    for (int c2 = 0; c2 < 16; c2++)
      s += q_lds[n * FLA_C + h * 16 + c2] * kv_lds[h * 256 + c2 * 16 + d];
    out_lds[i] = (__bf16)(s * rnum);
  }
  __syncthreads();

  // ---------------- Phase 5: GEMM3 (proj) via WMMA + scatter by voxel_inds ----------------
  // 24 tile-jobs (8 ntiles x 3 mtiles), 3 per wave
#pragma unroll
  for (int t = 0; t < 3; t++) {
    const int job = wave + t * 8;
    const int nt  = job & 7, mt = job >> 3;
    const __bf16* arow = out_lds + (mt * 16 + l16) * FLA_C;
    const __bf16* wp   = w_proj + (nt * 16 + l16) * FLA_C + g * 16;
    v8f c = {0.f, 0.f, 0.f, 0.f, 0.f, 0.f, 0.f, 0.f};
#pragma unroll
    for (int kb = 0; kb < 4; kb++) {
      v16bf af = make_afrag(arow + kb * 32, g);
      v16bf bf = *(const v16bf*)(wp + kb * 32);
      c = __builtin_amdgcn_wmma_f32_16x16x32_bf16(false, af, false, bf,
                                                  (short)0, c, false, false);
    }
    const int col = nt * 16 + l16;
    const float bias = proj_b[col];
#pragma unroll
    for (int r = 0; r < 8; r++) {
      const int m = mt * 16 + g * 8 + r;
      if (m < FLA_N) {
        const long long vox = voxel_inds[(size_t)b * FLA_N + m];
        out[(size_t)vox * FLA_C + col] = c[r] + bias;
      }
    }
  }
}

extern "C" void kernel_launch(void* const* d_in, const int* in_sizes, int n_in,
                              void* d_out, int out_size, void* d_ws, size_t ws_size,
                              hipStream_t stream) {
  (void)in_sizes; (void)n_in; (void)out_size; (void)ws_size;
  const float* x      = (const float*)d_in[0];
  const float* pos    = (const float*)d_in[1];
  const float* mask   = (const float*)d_in[2];
  const float* qk_w   = (const float*)d_in[3];
  const float* qk_b   = (const float*)d_in[4];
  const float* v_w    = (const float*)d_in[5];
  const float* v_b    = (const float*)d_in[6];
  const float* proj_w = (const float*)d_in[7];
  const float* proj_b = (const float*)d_in[8];
  /* d_in[9] = coords (unused by reference math) */
  const long long* voxel_inds = (const long long*)d_in[10];

  __bf16* wbf = (__bf16*)d_ws;   // needs 131072 bytes

  fla_prep_weights<<<256, 256, 0, stream>>>(qk_w, v_w, proj_w, wbf);
  fla_kernel<<<8192, 256, 0, stream>>>(x, pos, mask, qk_b, v_b, proj_b,
                                       voxel_inds, wbf, (float*)d_out);
}
